// TaxiCollectiveCriticWithCostAndBias_33328946217219
// MI455X (gfx1250) — matrix operand order repaired and checked by reading, not compiled, not run on hardware
//
#include <hip/hip_runtime.h>

typedef __attribute__((ext_vector_type(16))) _Float16 v16h;
typedef __attribute__((ext_vector_type(8)))  float    v8f;

#define ZN    81
#define HN    48
#define BN    8192
#define FEATN 210
#define FEATP 224   // padded to multiple of 32 for K-chunks
#define H1N   64
#define H2N   64
#define NSCALE 8000.0f

// ---------------------------------------------------------------------------
// Kernel A: streaming pass. One block per batch row b (96 threads).
//   - column reduction of action_count -> state_count, cost, cost_sum
//   - assemble padded f16 feature row [sc,served interleaved | time | 0-pad]
// ---------------------------------------------------------------------------
__global__ void prep_features_cost(const float* __restrict__ obs,
                                   const float* __restrict__ ac,
                                   const float* __restrict__ cm,
                                   _Float16* __restrict__ feat,
                                   float* __restrict__ sc_ws,
                                   float* __restrict__ cs_ws,
                                   float* __restrict__ cost_out) {
  const int b = blockIdx.x;
  const int t = threadIdx.x;
  const float* acb = ac + (size_t)b * ZN * ZN;
  float* cob = cost_out + (size_t)b * ZN * ZN;
  if (t < ZN) {
    float sc = 0.f, cs = 0.f;
    for (int i = 0; i < ZN; ++i) {
      float v = acb[i * ZN + t];            // coalesced across t
      float c = -v * cm[i * ZN + t];        // cost_matrix is L2-resident
      cob[i * ZN + t] = c;                  // coalesced store
      sc += v;
      cs += c;
    }
    float dc = obs[(size_t)b * ZN * 50 + t * 50 + 49];
    float served = fminf(sc, dc);
    feat[(size_t)b * FEATP + 2 * t]     = (_Float16)sc;
    feat[(size_t)b * FEATP + 2 * t + 1] = (_Float16)served;
    sc_ws[(size_t)b * ZN + t] = sc;
    cs_ws[(size_t)b * ZN + t] = cs;
  }
  if (t < HN) {
    float tp = obs[(size_t)b * ZN * 50 + t] * NSCALE;
    feat[(size_t)b * FEATP + 2 * ZN + t] = (_Float16)tp;
  }
  if (t < FEATP - FEATN) {
    feat[(size_t)b * FEATP + FEATN + t] = (_Float16)0.f;
  }
}

// ---------------------------------------------------------------------------
// Weight repack into WMMA B-fragment lane order (f32 -> f16).
// B-fragment (32x16, f16): element e of lane L holds
//   K = kc*32 + 16*(L/16) + e,  N = nt*16 + (L%16)
// stored contiguously so mlp_wmma loads one v16h (32B) per fragment per lane.
// ---------------------------------------------------------------------------
__global__ void prep_w1frag(const float* __restrict__ W1, _Float16* __restrict__ w1f) {
  const int blk  = blockIdx.x;        // blk = (z*4 + nt)*7 + kc
  const int kc   = blk % 7;
  const int nt   = (blk / 7) % 4;
  const int z    = blk / 28;
  const int lane = threadIdx.x;
  const int g = lane >> 4, ln = lane & 15;
  const int n = nt * 16 + ln;
  _Float16* out = w1f + (size_t)blk * 512 + lane * 16;
#pragma unroll
  for (int e = 0; e < 16; ++e) {
    int k = kc * 32 + g * 16 + e;
    float v = (k < FEATN) ? W1[((size_t)z * FEATN + k) * H1N + n] : 0.f;
    out[e] = (_Float16)v;
  }
}

__global__ void prep_w2frag(const float* __restrict__ W2, _Float16* __restrict__ w2f) {
  const int blk  = blockIdx.x;        // blk = (z*4 + nt)*2 + kc
  const int kc   = blk % 2;
  const int nt   = (blk / 2) % 4;
  const int z    = blk / 8;
  const int lane = threadIdx.x;
  const int g = lane >> 4, ln = lane & 15;
  const int n = nt * 16 + ln;
  _Float16* out = w2f + (size_t)blk * 512 + lane * 16;
#pragma unroll
  for (int e = 0; e < 16; ++e) {
    int k = kc * 32 + g * 16 + e;
    out[e] = (_Float16)W2[((size_t)z * H1N + k) * H2N + n];
  }
}

// ---------------------------------------------------------------------------
// Kernel B: one wave per (16-batch tile, zone).
//   GEMM1: 16x224 @ 224x64 (28 x v_wmma_f32_16x16x32_f16)
//   LDS relayout (C-layout -> A-layout), GEMM2: 16x64 @ 64x64 (8 WMMAs)
//   then f32 dot products + shuffle reduction for tw / d / state_returns.
// ---------------------------------------------------------------------------
__global__ void __launch_bounds__(32)
mlp_wmma(const _Float16* __restrict__ feat,
         const _Float16* __restrict__ w1f,
         const _Float16* __restrict__ w2f,
         const float* __restrict__ b1,
         const float* __restrict__ b2,
         const float* __restrict__ w_tw,
         const float* __restrict__ b_tw,
         const float* __restrict__ w_ub,
         const float* __restrict__ b_ub,
         const float* __restrict__ sc_ws,
         const float* __restrict__ cs_ws,
         float* __restrict__ out_sr,
         float* __restrict__ out_d,
         float* __restrict__ out_tw) {
  const int z    = blockIdx.y;
  const int m0   = blockIdx.x * 16;
  const int lane = threadIdx.x;
  const int g = lane >> 4, ln = lane & 15;

  __shared__ _Float16 hsm[16 * 68];   // 68-half row stride: conflict-free TR reads

  union Frag { v16h v; unsigned int u[8]; };

  // ---- A fragments: feature tile 16 x 224 (f16). Pair p of lane L covers
  //      K = kc*32 + 16*(L/16) + 2*p (+0,+1): one dword load. ----
  Frag afr[7];
  const _Float16* frow = feat + (size_t)(m0 + ln) * FEATP;
#pragma unroll
  for (int kc = 0; kc < 7; ++kc) {
#pragma unroll
    for (int p = 0; p < 8; ++p) {
      int kp = kc * 32 + 16 * g + 2 * p;
      afr[kc].u[p] = *(const unsigned int*)(frow + kp);
    }
  }

  // ---- GEMM1 + bias + relu -> LDS (f16) ----
#pragma unroll
  for (int nt = 0; nt < 4; ++nt) {
    v8f acc = {0.f, 0.f, 0.f, 0.f, 0.f, 0.f, 0.f, 0.f};
#pragma unroll
    for (int kc = 0; kc < 7; ++kc) {
      const v16h bfrag =
          *(const v16h*)(w1f + (size_t)((z * 4 + nt) * 7 + kc) * 512 + lane * 16);
      acc = __builtin_amdgcn_wmma_f32_16x16x32_f16(
          false, afr[kc].v, false, bfrag, (short)0, acc, false, false);
    }
    float bias = b1[z * H1N + nt * 16 + ln];
#pragma unroll
    for (int r = 0; r < 8; ++r) {         // C-layout: row m = r + 8*(lane/16)
      int m = r + 8 * g;
      float hv = acc[r] + bias;
      hv = hv > 0.f ? hv : 0.f;
      hsm[m * 68 + nt * 16 + ln] = (_Float16)hv;
    }
  }
  __syncthreads();

  // ---- reload h1 in A-fragment layout ----
  Frag a2[2];
#pragma unroll
  for (int kc = 0; kc < 2; ++kc) {
#pragma unroll
    for (int p = 0; p < 8; ++p) {
      int kp = kc * 32 + 16 * g + 2 * p;
      a2[kc].u[p] = *(const unsigned int*)(&hsm[ln * 68 + kp]);
    }
  }

  // ---- GEMM2 + bias + relu + per-row dot accumulation ----
  float tw_acc[8], d_acc[8];
#pragma unroll
  for (int r = 0; r < 8; ++r) { tw_acc[r] = 0.f; d_acc[r] = 0.f; }

#pragma unroll
  for (int nt = 0; nt < 4; ++nt) {
    v8f acc = {0.f, 0.f, 0.f, 0.f, 0.f, 0.f, 0.f, 0.f};
#pragma unroll
    for (int kc = 0; kc < 2; ++kc) {
      const v16h bfrag =
          *(const v16h*)(w2f + (size_t)((z * 4 + nt) * 2 + kc) * 512 + lane * 16);
      acc = __builtin_amdgcn_wmma_f32_16x16x32_f16(
          false, a2[kc].v, false, bfrag, (short)0, acc, false, false);
    }
    int n = nt * 16 + ln;
    float bias = b2[z * H2N + n];
    float wtw  = w_tw[z * H2N + n];
    float wub  = w_ub[z * H2N + n];
#pragma unroll
    for (int r = 0; r < 8; ++r) {
      float hv = acc[r] + bias;
      hv = hv > 0.f ? hv : 0.f;
      tw_acc[r] += hv * wtw;
      d_acc[r]  += hv * wub;
    }
  }

  // reduce over the 16 N-lanes of each half-wave (masks stay within half)
#pragma unroll
  for (int mask = 1; mask <= 8; mask <<= 1) {
#pragma unroll
    for (int r = 0; r < 8; ++r) {
      tw_acc[r] += __shfl_xor(tw_acc[r], mask, 32);
      d_acc[r]  += __shfl_xor(d_acc[r],  mask, 32);
    }
  }

  if (ln < 8) {                      // lane ln of half g owns row m = ln + 8*g
    float twv = 0.f, dvv = 0.f;
#pragma unroll
    for (int r = 0; r < 8; ++r) {
      if (ln == r) { twv = tw_acc[r]; dvv = d_acc[r]; }
    }
    int b = m0 + ln + 8 * g;
    twv += b_tw[z];
    dvv += b_ub[z];
    float sc = sc_ws[(size_t)b * ZN + z];
    float cs = cs_ws[(size_t)b * ZN + z];
    float sr = twv * fminf(sc, dvv) + cs;
    out_tw[(size_t)b * ZN + z] = twv;
    out_d [(size_t)b * ZN + z] = dvv;
    out_sr[(size_t)b * ZN + z] = sr;
  }
}

// ---------------------------------------------------------------------------
// Kernel C: symbolic_val[b] = sum_z state_returns[b][z]
// ---------------------------------------------------------------------------
__global__ void reduce_sym(const float* __restrict__ sr, float* __restrict__ sym) {
  int b = blockIdx.x * blockDim.x + threadIdx.x;
  if (b < BN) {
    float s = 0.f;
    for (int zz = 0; zz < ZN; ++zz) s += sr[(size_t)b * ZN + zz];
    sym[b] = s;
  }
}

// ---------------------------------------------------------------------------
extern "C" void kernel_launch(void* const* d_in, const int* in_sizes, int n_in,
                              void* d_out, int out_size, void* d_ws, size_t ws_size,
                              hipStream_t stream) {
  (void)in_sizes; (void)n_in; (void)out_size; (void)ws_size;

  const float* obs  = (const float*)d_in[0];
  const float* ac   = (const float*)d_in[1];
  const float* cm   = (const float*)d_in[2];
  const float* W1   = (const float*)d_in[3];
  const float* b1   = (const float*)d_in[4];
  const float* W2   = (const float*)d_in[5];
  const float* b2   = (const float*)d_in[6];
  const float* w_tw = (const float*)d_in[7];
  const float* b_tw = (const float*)d_in[8];
  const float* w_ub = (const float*)d_in[9];
  const float* b_ub = (const float*)d_in[10];

  // Output tuple layout: sym(B) | sr(B*Z) | cost(B*Z*Z) | d(B*Z) | tw(B*Z)
  float* out  = (float*)d_out;
  float* sym  = out;
  float* sr   = out + BN;
  float* cost = sr + (size_t)BN * ZN;
  float* dd   = cost + (size_t)BN * ZN * ZN;
  float* tw   = dd + (size_t)BN * ZN;

  // Workspace layout (all 32B aligned)
  char* ws = (char*)d_ws;
  _Float16* feat = (_Float16*)(ws);                                  // B*224 f16
  _Float16* w1f  = (_Float16*)(ws + (size_t)3670016);                // 81*28*512 f16
  _Float16* w2f  = (_Float16*)(ws + (size_t)5992448);                // 81*8*512 f16
  float*    scw  = (float*)   (ws + (size_t)6656000);                // B*Z f32
  float*    csw  = (float*)   (ws + (size_t)9310208);                // B*Z f32

  prep_features_cost<<<BN, 96, 0, stream>>>(obs, ac, cm, feat, scw, csw, cost);
  prep_w1frag<<<ZN * 4 * 7, 32, 0, stream>>>(W1, w1f);
  prep_w2frag<<<ZN * 4 * 2, 32, 0, stream>>>(W2, w2f);

  dim3 grid(BN / 16, ZN);
  mlp_wmma<<<grid, 32, 0, stream>>>(feat, w1f, w2f, b1, b2, w_tw, b_tw, w_ub, b_ub,
                                    scw, csw, sr, dd, tw);

  reduce_sym<<<BN / 256, 256, 0, stream>>>(sr, sym);
}
